// Repro_28913719837028
// MI455X (gfx1250) — matrix operand order repaired and checked by reading, not compiled, not run on hardware
//
#include <hip/hip_runtime.h>

#define N_HEAD 16
#define D_HEAD 64
#define D_MODEL 1024
#define SEQ 3072

typedef __attribute__((ext_vector_type(16))) _Float16 v16h;
typedef __attribute__((ext_vector_type(8)))  _Float16 v8h;
typedef __attribute__((ext_vector_type(8)))  float    v8f;

#if defined(__has_builtin)
#  if __has_builtin(__builtin_amdgcn_tensor_load_to_lds) && \
      __has_builtin(__builtin_amdgcn_s_wait_tensorcnt)
#    define HAVE_TDM 1
#  endif
#endif
#ifndef HAVE_TDM
#  define HAVE_TDM 0
#endif

static __device__ __forceinline__ v16h pack16(v8h lo, v8h hi) {
    v16h r;
#pragma unroll
    for (int i = 0; i < 8; ++i) { r[i] = lo[i]; r[8 + i] = hi[i]; }
    return r;
}

#if HAVE_TDM
typedef __attribute__((ext_vector_type(4))) unsigned int tdm_u32x4;
typedef __attribute__((ext_vector_type(8))) int          tdm_i32x8;
typedef __attribute__((ext_vector_type(4))) int          tdm_i32x4;

// Issue a TDM load of a 16-key x 64-half (2KB) K tile into LDS.
// D# built per CDNA5 ISA §8.3/8.4: 2D tensor, data_size=2B,
// tensor_dim0=64 (contiguous d), tensor_dim1=16, stride0=64,
// tile_dim0=64, tile_dim1=16.
static __device__ __forceinline__ void tdm_load_ktile(
    const _Float16* gsrc, unsigned int lds_off)
{
    unsigned long long ga = (unsigned long long)(size_t)gsrc;
    tdm_u32x4 g0;
    g0[0] = 1u;                                   // count=1, user descriptor
    g0[1] = lds_off;                              // lds_addr (bytes)
    g0[2] = (unsigned int)ga;                     // global_addr[31:0]
    g0[3] = (unsigned int)((ga >> 32) & 0x1FFFFFFull) | (2u << 30); // [56:32]+type=2
    tdm_i32x8 g1;
    g1[0] = (1 << 16);                            // wg_mask=0, data_size=1 (2 bytes)
    g1[1] = (int)((64u & 0xFFFFu) << 16);         // tensor_dim0[15:0] in bits 63:48
    g1[2] = (int)((64u >> 16) | ((16u & 0xFFFFu) << 16)); // dim0 hi | tensor_dim1 lo
    g1[3] = (int)((16u >> 16) | (64u << 16));     // tensor_dim1 hi | tile_dim0=64
    g1[4] = (int)16u;                             // tile_dim1=16, tile_dim2=0
    g1[5] = (int)64u;                             // tensor_dim0_stride[31:0]=64
    g1[6] = 0;                                    // stride0 hi | dim1_stride lo
    g1[7] = 0;
    tdm_i32x4 z4 = {0, 0, 0, 0};
#if __clang_major__ >= 23
    tdm_i32x8 z8 = {0, 0, 0, 0, 0, 0, 0, 0};
    __builtin_amdgcn_tensor_load_to_lds(g0, g1, z4, z4, z8, 0);
#else
    __builtin_amdgcn_tensor_load_to_lds(g0, g1, z4, z4, 0);
#endif
}
#endif  // HAVE_TDM

// ---------------------------------------------------------------------------
// Kernel 1: qkv = x @ Wqkv^T + bqkv -> f16 per-head tensors:
//   Qh[h][s][d] (pre-scaled by 1/8), Kh[h][s][d], Vt[h][d][s].
// One wave computes a 16x64 output tile (4 accumulators) so the A fragment
// is reused across 4 WMMAs per k-step. The 64-col group == one head, so the
// epilogue region/head selection is wave-uniform.
// ---------------------------------------------------------------------------
__global__ __launch_bounds__(256) void qkv_proj_wmma(
    const float* __restrict__ x, const float* __restrict__ Wqkv,
    const float* __restrict__ bqkv,
    _Float16* __restrict__ Qh, _Float16* __restrict__ Kh,
    _Float16* __restrict__ Vt)
{
    const int wid  = blockIdx.x * 8 + (threadIdx.x >> 5);
    const int lane = threadIdx.x & 31;
    const int ln = lane & 15;
    const int kh = lane >> 4;

    const int mblk = wid % (SEQ / 16);        // 192 row tiles (tokens)
    const int n4   = wid / (SEQ / 16);        // 48 tiles of 64 output columns
    const int mrow = mblk * 16;

    const float* xr  = x + (size_t)(mrow + ln) * D_MODEL;           // A row m = ln
    const float* wr0 = Wqkv + (size_t)(n4 * 64 + ln) * D_MODEL;     // B col base

    v8f acc[4] = {};
    for (int kb = 0; kb < D_MODEL; kb += 32) {
        __builtin_prefetch(xr + kb + 64, 0, 0);
        v16h a;
        // A fragment (16x32 f16): lanes 0-15 -> K {kb..kb+7, kb+16..kb+23},
        // lanes 16-31 -> K {kb+8..kb+15, kb+24..kb+31}
#pragma unroll
        for (int e = 0; e < 8; ++e) {
            a[e]     = (_Float16)xr[kb + kh * 8 + e];
            a[8 + e] = (_Float16)xr[kb + 16 + kh * 8 + e];
        }
#pragma unroll
        for (int j = 0; j < 4; ++j) {
            const float* wr = wr0 + (size_t)j * 16 * D_MODEL;
            v16h b;
#pragma unroll
            for (int e = 0; e < 16; ++e)
                b[e] = (_Float16)wr[kb + ((e >> 3) << 4) + kh * 8 + (e & 7)];
            acc[j] = __builtin_amdgcn_wmma_f32_16x16x32_f16(
                false, a, false, b, (short)0, acc[j], false, false);
        }
    }

    const int region = n4 >> 4;        // 0=Q, 1=K, 2=V (wave-uniform)
    const int hh     = n4 & 15;        // head (wave-uniform)
#pragma unroll
    for (int j = 0; j < 4; ++j) {
        const int ncol = n4 * 64 + j * 16 + ln;
        const int dd   = j * 16 + ln;
        const float bv = bqkv[ncol];
#pragma unroll
        for (int r = 0; r < 8; ++r) {
            const int s = mrow + r + kh * 8;   // C/D layout: M = r + 8*laneHalf
            const float v = acc[j][r] + bv;
            if (region == 0)
                Qh[((size_t)hh * SEQ + s) * D_HEAD + dd] = (_Float16)(v * 0.125f);
            else if (region == 1)
                Kh[((size_t)hh * SEQ + s) * D_HEAD + dd] = (_Float16)v;
            else
                Vt[((size_t)hh * D_HEAD + dd) * SEQ + s] = (_Float16)v;
        }
    }
}

// ---------------------------------------------------------------------------
// Kernel 2: attention for one (head, 16-query block) per workgroup.
// 4 waves, 128 threads. LDS holds the full 16 x 3072 f32 score stripe
// (192KB -- only possible with CDNA5's 320KB LDS/WGP), so softmax is a
// simple one-pass, no online rescaling. K tiles are staged into LDS by the
// Tensor Data Mover (double-buffered, s_wait_tensorcnt pipelined).
// ---------------------------------------------------------------------------
__global__ __launch_bounds__(128) void attn_wmma(
    const _Float16* __restrict__ Qh, const _Float16* __restrict__ Kh,
    const _Float16* __restrict__ Vt, float* __restrict__ out)
{
    extern __shared__ float smem[];
    float* sS   = smem;                  // [16][SEQ] scores / probs
    float* red  = smem + 16 * SEQ;       // [128] reduction scratch
    float* rmax = red + 128;             // [16]
    float* rinv = rmax + 16;             // [16]
    _Float16* kst = (_Float16*)(rinv + 16);  // 4 waves x 2 bufs x 1024 halves

    const int tid  = threadIdx.x;
    const int wv   = tid >> 5;           // wave 0..3
    const int lane = tid & 31;
    const int ln = lane & 15;
    const int kh = lane >> 4;

    const int head = blockIdx.x / (SEQ / 16);
    const int qblk = blockIdx.x % (SEQ / 16);

    // Preload Q A-fragments for both 32-wide d chunks (row m = ln)
    const _Float16* qrow = Qh + ((size_t)head * SEQ + qblk * 16 + ln) * D_HEAD;
    v16h qa[2];
#pragma unroll
    for (int c = 0; c < 2; ++c) {
        v8h lo = *(const v8h*)(qrow + c * 32 + kh * 8);
        v8h hi = *(const v8h*)(qrow + c * 32 + kh * 8 + 16);
        qa[c] = pack16(lo, hi);
    }

    // q-block grid coordinates (constant over the 16-row block except Hq)
    const float Fq = (float)(qblk >> 4);
    const float Wq = (float)(qblk & 15);

    // ---- Phase A: score tiles -> LDS -------------------------------------
#if HAVE_TDM
    _Float16* myk = kst + (size_t)wv * 2048;                 // 2 bufs x 2KB
    const unsigned koff = (unsigned)(size_t)(void*)myk;      // LDS byte offset
    tdm_load_ktile(Kh + ((size_t)head * SEQ + wv * 16) * D_HEAD, koff);
    int buf = 0;
#endif
    for (int kvt = wv; kvt < SEQ / 16; kvt += 4) {
        const int ktok = kvt * 16 + ln;                      // B column n = ln
        v8f c = {};
#if HAVE_TDM
        const int nxt = kvt + 4;
        if (nxt < SEQ / 16) {
            tdm_load_ktile(Kh + ((size_t)head * SEQ + nxt * 16) * D_HEAD,
                           koff + (unsigned)((buf ^ 1) * 2048));
            __builtin_amdgcn_s_wait_tensorcnt(1);
        } else {
            __builtin_amdgcn_s_wait_tensorcnt(0);
        }
        const _Float16* krow = myk + buf * 1024 + ln * D_HEAD;  // LDS tile row
        buf ^= 1;
#else
        const _Float16* krow = Kh + ((size_t)head * SEQ + ktok) * D_HEAD;
#endif
#pragma unroll
        for (int ch = 0; ch < 2; ++ch) {
            v8h lo = *(const v8h*)(krow + ch * 32 + kh * 8);
            v8h hi = *(const v8h*)(krow + ch * 32 + kh * 8 + 16);
            v16h b = pack16(lo, hi);
            c = __builtin_amdgcn_wmma_f32_16x16x32_f16(
                false, qa[ch], false, b, (short)0, c, false, false);
        }
        // Analytic Brownian bias (both factors use 0.2 => collapsed form)
        const float Fk = (float)(ktok >> 8);
        const float Wk = (float)((ktok >> 4) & 15);
        const float Hk = (float)(ktok & 15);
        const float df  = fabsf(Fq - Fk);
        const float rct = 20.0f / fmaxf(df, 1.0f);           // 1/ct
        const float coef = 3.9788735772973836f * rct;        // 1/(2*pi*0.04*ct)
        const float dwq = Wq - Wk;
        const float dw2 = dwq * dwq;
        const float g = 12.5f * rct;                         // 1/(0.08*ct)
#pragma unroll
        for (int r = 0; r < 8; ++r) {
            const float dh = (float)(r + kh * 8) - Hk;       // Hq - Hk
            const float bias = coef * __expf(-(dw2 + dh * dh) * g);
            sS[(r + kh * 8) * SEQ + ktok] = c[r] + bias;
        }
    }
    __syncthreads();

    // ---- Phase B: row softmax in LDS -------------------------------------
    const int row = tid >> 3;            // 0..15
    const int seg = tid & 7;             // 0..7, each covers SEQ/8 = 384
    float* srow = sS + row * SEQ;
    const int j0 = seg * (SEQ / 8);
    float lmax = -3.0e38f;
    for (int j = j0; j < j0 + SEQ / 8; ++j) lmax = fmaxf(lmax, srow[j]);
    red[tid] = lmax;
    __syncthreads();
    if (tid < 16) {
        float m = -3.0e38f;
#pragma unroll
        for (int i = 0; i < 8; ++i) m = fmaxf(m, red[tid * 8 + i]);
        rmax[tid] = m;
    }
    __syncthreads();
    const float rm = rmax[row];
    float lsum = 0.0f;
    for (int j = j0; j < j0 + SEQ / 8; ++j) {
        const float e = __expf(srow[j] - rm);
        srow[j] = e;
        lsum += e;
    }
    red[tid] = lsum;
    __syncthreads();
    if (tid < 16) {
        float s = 0.0f;
#pragma unroll
        for (int i = 0; i < 8; ++i) s += red[tid * 8 + i];
        rinv[tid] = 1.0f / s;
    }
    __syncthreads();

    // ---- Phase C: out = P @ V, wave wv owns d columns [wv*16, wv*16+16) ---
    const int dcol = wv * 16 + ln;                           // B/C column n = ln
    const _Float16* vrow = Vt + ((size_t)head * D_HEAD + dcol) * SEQ;
    const float* arow = sS + ln * SEQ;                       // A row m = ln
    const float ainv = rinv[ln];
    v8f acc = {};
    for (int kc = 0; kc < SEQ / 32; ++kc) {
        const int kb = kc * 32;
        __builtin_prefetch(vrow + kb + 64, 0, 0);
        v16h a;
#pragma unroll
        for (int e = 0; e < 8; ++e) {
            a[e]     = (_Float16)(arow[kb + kh * 8 + e] * ainv);
            a[8 + e] = (_Float16)(arow[kb + 16 + kh * 8 + e] * ainv);
        }
        v8h lo = *(const v8h*)(vrow + kb + kh * 8);
        v8h hi = *(const v8h*)(vrow + kb + kh * 8 + 16);
        v16h b = pack16(lo, hi);
        acc = __builtin_amdgcn_wmma_f32_16x16x32_f16(
            false, a, false, b, (short)0, acc, false, false);
    }
#pragma unroll
    for (int r = 0; r < 8; ++r) {
        const int qtok = qblk * 16 + r + kh * 8;
        out[((size_t)head * SEQ + qtok) * D_HEAD + dcol] = acc[r];
    }
}

// ---------------------------------------------------------------------------
extern "C" void kernel_launch(void* const* d_in, const int* in_sizes, int n_in,
                              void* d_out, int out_size, void* d_ws, size_t ws_size,
                              hipStream_t stream) {
    const float* x    = (const float*)d_in[0];
    const float* Wqkv = (const float*)d_in[1];
    const float* bqkv = (const float*)d_in[2];
    float* out = (float*)d_out;

    _Float16* Qh = (_Float16*)d_ws;                              // 6 MB
    _Float16* Kh = Qh + (size_t)N_HEAD * SEQ * D_HEAD;           // 6 MB
    _Float16* Vt = Kh + (size_t)N_HEAD * SEQ * D_HEAD;           // 6 MB

    // Kernel 1: 192 x 48 output tiles (16x64 each), 8 waves per 256-thr block
    const int tiles = (SEQ / 16) * (3 * D_MODEL / 64);           // 9216
    qkv_proj_wmma<<<tiles / 8, 256, 0, stream>>>(x, Wqkv, bqkv, Qh, Kh, Vt);

    // Kernel 2: one workgroup per (head, 16-query block)
    // LDS: 192KB scores + reductions + 16KB TDM K-tile staging (~210KB)
    const size_t smemBytes =
        (size_t)(16 * SEQ + 128 + 16 + 16) * sizeof(float) +
        (size_t)4 * 2 * 1024 * sizeof(_Float16);
    attn_wmma<<<N_HEAD * (SEQ / 16), 128, smemBytes, stream>>>(Qh, Kh, Vt, out);
}
